// LSTMDecoder_65730179498210
// MI455X (gfx1250) — matrix operand order repaired and checked by reading, not compiled, not run on hardware
//
#include <hip/hip_runtime.h>

#define Hdim 1024
#define Ndim 1024
#define Ldim 96
#define EPSV 1e-5f

typedef __bf16 bf16;
typedef bf16  v16bf __attribute__((ext_vector_type(16)));
typedef float v8f   __attribute__((ext_vector_type(8)));
typedef float v4f   __attribute__((ext_vector_type(4)));

union FragBF {
  v16bf v;
  uint4 q[2];
};

struct Frags {
  FragBF a0, a1;   // two 16-row A fragments (rows [0..15], [16..31] of wave tile)
  FragBF b[4];     // one B fragment per gate (i, f, g, o)
};

__device__ __forceinline__ unsigned short f2bf(float f) {
  unsigned int u = __float_as_uint(f);
  u += 0x7FFFu + ((u >> 16) & 1u);   // round-to-nearest-even
  return (unsigned short)(u >> 16);
}

__device__ __forceinline__ float fsigmoid(float x) {
  return 1.0f / (1.0f + __expf(-x));
}

__device__ __forceinline__ float ftanh(float x) {
  // tanh(x) = 1 - 2/(exp(2x)+1); saturates correctly at +/-1
  return 1.0f - 2.0f / (__expf(2.0f * x) + 1.0f);
}

// One-time per-call prep: W fp32 -> bf16, h_in fp32 -> bf16, c = 0.
__global__ void prep_kernel(const float* __restrict__ h_in,
                            const float* __restrict__ W,
                            unsigned short* __restrict__ Wbf,
                            unsigned short* __restrict__ h0,
                            float* __restrict__ c) {
  int idx = blockIdx.x * blockDim.x + threadIdx.x;
  if (idx < 4 * Hdim * Hdim) Wbf[idx] = f2bf(W[idx]);
  if (idx < Ndim * Hdim) {
    h0[idx] = f2bf(h_in[idx]);
    c[idx] = 0.0f;
  }
}

// One LSTM step: gates = h @ W^T + b; update c,h; emit h to out[:, :, l];
// emit deterministic per-(block,channel) partial sum/sumsq for BatchNorm.
// Block: 256 threads = 8 waves. Block tile: 256 rows (N) x 16 cols (H).
// Each wave: 32 rows x 16 cols, 4 gates -> 8 WMMA accumulators (v8f each).
// K loop is double-buffered (f0/f1) so each WMMA group hides the other
// buffer's global_load_b128 latency.
__global__ __launch_bounds__(256) void lstm_step_kernel(
    const unsigned short* __restrict__ hprev,   // (N,H) bf16
    const unsigned short* __restrict__ Wbf,     // (4H,H) bf16
    const float* __restrict__ b_ih,
    const float* __restrict__ b_hh,
    float* __restrict__ c,                      // (N,H) f32
    unsigned short* __restrict__ hnext,         // (N,H) bf16
    float* __restrict__ out,                    // (N,H,L) f32
    float* __restrict__ psum,                   // (L*4, H) partials
    float* __restrict__ psq,
    int l) {
  const int tid  = threadIdx.x;
  const int lane = tid & 31;
  const int wave = tid >> 5;
  const int m16  = lane & 15;
  const bool lo  = (lane < 16);
  const int col0 = blockIdx.x * 16;
  const int row0 = blockIdx.y * 256 + wave * 32;
  const int col  = col0 + m16;

  const v8f vzero = {0.f, 0.f, 0.f, 0.f, 0.f, 0.f, 0.f, 0.f};
  v8f acc[2][4] = {{vzero, vzero, vzero, vzero}, {vzero, vzero, vzero, vzero}};

  // A fragment (16x32 bf16, M x K): lane = row M; lo half holds K {0..7,16..23},
  // hi half holds K {8..15,24..31} of the current K-chunk.
  const unsigned short* aBase0 = hprev + (size_t)(row0 + m16) * Hdim;
  const unsigned short* aBase1 = hprev + (size_t)(row0 + 16 + m16) * Hdim;
  const int aoff = lo ? 0 : 8;
  // B fragment (32x16 bf16, K x N): lane = column N; lo half K {0..7,16..23}, hi {8..15,24..31}.
  // B = W^T, so column N = W row (g*H + col), K-contiguous in memory.
  const int boff = lo ? 0 : 8;
  const unsigned short* bBase[4];
#pragma unroll
  for (int g = 0; g < 4; ++g) {
    bBase[g] = Wbf + (size_t)(g * Hdim + col) * Hdim + boff;
  }

  auto load_frags = [&](Frags& F, int kk) {
    F.a0.q[0] = *(const uint4*)(aBase0 + kk + aoff);
    F.a0.q[1] = *(const uint4*)(aBase0 + kk + 16 + aoff);
    F.a1.q[0] = *(const uint4*)(aBase1 + kk + aoff);
    F.a1.q[1] = *(const uint4*)(aBase1 + kk + 16 + aoff);
#pragma unroll
    for (int g = 0; g < 4; ++g) {
      F.b[g].q[0] = *(const uint4*)(bBase[g] + kk);
      F.b[g].q[1] = *(const uint4*)(bBase[g] + kk + 16);
    }
  };

  auto do_mma = [&](Frags& F) {
#pragma unroll
    for (int g = 0; g < 4; ++g) {
      acc[0][g] = __builtin_amdgcn_wmma_f32_16x16x32_bf16(
          false, F.a0.v, false, F.b[g].v, (short)0, acc[0][g], false, false);
      acc[1][g] = __builtin_amdgcn_wmma_f32_16x16x32_bf16(
          false, F.a1.v, false, F.b[g].v, (short)0, acc[1][g], false, false);
    }
  };

  Frags f0, f1;
  load_frags(f0, 0);
  for (int kk = 0; kk < Hdim; kk += 64) {
    load_frags(f1, kk + 32);
    __builtin_prefetch(aBase0 + kk + 128, 0, 3);
    do_mma(f0);
    if (kk + 64 < Hdim) load_frags(f0, kk + 64);
    do_mma(f1);
  }

  // Per-column gate biases (same for all accumulator rows of this lane).
  const float bi = b_ih[col]            + b_hh[col];
  const float bf = b_ih[Hdim + col]     + b_hh[Hdim + col];
  const float bg = b_ih[2 * Hdim + col] + b_hh[2 * Hdim + col];
  const float bo = b_ih[3 * Hdim + col] + b_hh[3 * Hdim + col];

  // C/D layout: element r -> (M = r + (hi?8:0), N = lane%16).
  const int rowOff = lo ? 0 : 8;
  float s = 0.0f, sq = 0.0f;
#pragma unroll
  for (int p = 0; p < 2; ++p) {
    const int rbase = row0 + p * 16 + rowOff;
#pragma unroll
    for (int r = 0; r < 8; ++r) {
      const int row = rbase + r;
      const size_t idx = (size_t)row * Hdim + col;
      const float iv = fsigmoid(acc[p][0][r] + bi);
      const float fv = fsigmoid(acc[p][1][r] + bf);
      const float gv = ftanh(acc[p][2][r] + bg);
      const float ov = fsigmoid(acc[p][3][r] + bo);
      const float cn = fv * c[idx] + iv * gv;
      c[idx] = cn;
      const float hv = ov * ftanh(cn);
      hnext[idx] = f2bf(hv);
      // Streaming 402MB output: non-temporal so W/h/c stay resident in L2.
      __builtin_nontemporal_store(
          hv, &out[(size_t)row * (Hdim * Ldim) + (size_t)col * Ldim + l]);
      s += hv;
      sq += hv * hv;
    }
  }

  // Deterministic per-channel block reduction (16 channels per block).
  __shared__ float red_s[256];
  __shared__ float red_q[256];
  red_s[tid] = s;
  red_q[tid] = sq;
  __syncthreads();
  if (tid < 16) {
    float ts = 0.0f, tq = 0.0f;
#pragma unroll
    for (int w = 0; w < 8; ++w) {
      ts += red_s[w * 32 + tid] + red_s[w * 32 + tid + 16];
      tq += red_q[w * 32 + tid] + red_q[w * 32 + tid + 16];
    }
    const int slot = (l * 4 + blockIdx.y) * Hdim + col0 + tid;
    psum[slot] = ts;
    psq[slot]  = tq;
  }
}

// Fold partials -> per-channel scale/shift for BatchNorm (training-mode stats).
__global__ void stats_kernel(const float* __restrict__ psum,
                             const float* __restrict__ psq,
                             const float* __restrict__ gamma,
                             const float* __restrict__ beta,
                             float* __restrict__ scale,
                             float* __restrict__ shift) {
  const int ch = blockIdx.x * blockDim.x + threadIdx.x;
  if (ch >= Hdim) return;
  float s = 0.0f, q = 0.0f;
  for (int j = 0; j < Ldim * 4; ++j) {
    s += psum[(size_t)j * Hdim + ch];
    q += psq[(size_t)j * Hdim + ch];
  }
  const float inv = 1.0f / (float)(Ndim * Ldim);
  const float mean = s * inv;
  const float var = q * inv - mean * mean;
  const float rs = rsqrtf(var + EPSV);
  const float sc = gamma[ch] * rs;
  scale[ch] = sc;
  shift[ch] = beta[ch] - mean * sc;
}

// Streaming in-place normalize: out = out*scale[ch] + shift[ch].
// Vectorized 4-wide along L (L = 96 = 24 * 4, 16B aligned); fully NT.
__global__ void norm_kernel(float* __restrict__ out,
                            const float* __restrict__ scale,
                            const float* __restrict__ shift) {
  const int idx = blockIdx.x * blockDim.x + threadIdx.x;  // over N*H*24
  const int nch = idx / 24;
  const int l4 = idx - nch * 24;
  const int ch = nch & (Hdim - 1);
  v4f* p = (v4f*)out + (size_t)nch * 24 + l4;
  v4f v = __builtin_nontemporal_load(p);
  const float sc = scale[ch];
  const float sh = shift[ch];
#pragma unroll
  for (int e = 0; e < 4; ++e) v[e] = v[e] * sc + sh;
  __builtin_nontemporal_store(v, p);
}

extern "C" void kernel_launch(void* const* d_in, const int* in_sizes, int n_in,
                              void* d_out, int out_size, void* d_ws,
                              size_t ws_size, hipStream_t stream) {
  const float* h_in  = (const float*)d_in[0];
  const float* W     = (const float*)d_in[1];
  const float* b_ih  = (const float*)d_in[2];
  const float* b_hh  = (const float*)d_in[3];
  const float* gamma = (const float*)d_in[4];
  const float* beta  = (const float*)d_in[5];
  float* out = (float*)d_out;

  // Workspace layout (bytes):
  //   [0,  8 MiB)  W bf16 (4H*H)
  //   [8 MiB)      h ping bf16 (N*H)
  //   [10 MiB)     h pong bf16 (N*H)
  //   [12 MiB)     c f32 (N*H)
  //   [16 MiB)     psum (L*4*H f32)
  //   [17.5 MiB)   psq  (L*4*H f32)
  //   [19 MiB)     scale[H], shift[H]
  char* ws = (char*)d_ws;
  unsigned short* Wbf = (unsigned short*)(ws + 0);
  unsigned short* h0  = (unsigned short*)(ws + 8388608);
  unsigned short* h1  = (unsigned short*)(ws + 10485760);
  float* c     = (float*)(ws + 12582912);
  float* psum  = (float*)(ws + 16777216);
  float* psq   = (float*)(ws + 18350080);
  float* scale = (float*)(ws + 19922944);
  float* shift = (float*)(ws + 19927040);

  prep_kernel<<<(4 * Hdim * Hdim + 255) / 256, 256, 0, stream>>>(h_in, W, Wbf,
                                                                 h0, c);

  dim3 sgrid(Hdim / 16, Ndim / 256);
  for (int l = 0; l < Ldim; ++l) {
    const unsigned short* hp = (l & 1) ? h1 : h0;
    unsigned short* hn = (l & 1) ? h0 : h1;
    lstm_step_kernel<<<sgrid, 256, 0, stream>>>(hp, Wbf, b_ih, b_hh, c, hn, out,
                                                psum, psq, l);
  }

  stats_kernel<<<(Hdim + 255) / 256, 256, 0, stream>>>(psum, psq, gamma, beta,
                                                       scale, shift);
  norm_kernel<<<(Ndim * Hdim * (Ldim / 4)) / 256, 256, 0, stream>>>(out, scale,
                                                                    shift);
}
